// ImageTransformer2D_3D_PETR_49435073577586
// MI455X (gfx1250) — matrix operand order repaired and checked by reading, not compiled, not run on hardware
//
#include <hip/hip_runtime.h>
#include <hip/hip_bf16.h>
#include <cstdint>

typedef __attribute__((ext_vector_type(16))) _Float16 v16h;
typedef __attribute__((ext_vector_type(8)))  _Float16 v8h;
typedef __attribute__((ext_vector_type(8)))  float    v8f;

#define C_DIM   128
#define N_Q     200
#define MQ_PAD  208
#define H_N     8
#define V_N     6
#define K_PV    5600
#define VK_N    33600
#define F_DIM   256
#define IMG_XC  100.0f
#define IMG_YC  56.0f
#define OUT_SFC 8.0f
#define EPS32F  1.1920929e-7f
#define NEG_BIG -1e30f

__device__ __forceinline__ float sigmf(float x) { return 1.0f / (1.0f + __expf(-x)); }

__device__ __forceinline__ v16h load_a16(const _Float16* p0, const _Float16* p1) {
  v8h a0 = *(const v8h*)p0;
  v8h a1 = *(const v8h*)p1;
  v16h r;
#pragma unroll
  for (int i = 0; i < 8; ++i) { r[i] = a0[i]; r[i + 8] = a1[i]; }
  return r;
}
__device__ __forceinline__ v16h zero16() {
  v16h r;
#pragma unroll
  for (int i = 0; i < 16; ++i) r[i] = (_Float16)0.0f;
  return r;
}
__device__ __forceinline__ v8f zero8f() {
  v8f r;
#pragma unroll
  for (int i = 0; i < 8; ++i) r[i] = 0.0f;
  return r;
}

// ---------------------------------------------------------------------------
// Generic WMMA GEMM:  out[M,Nout] = A[M,Ka](f16) * W^T  (W stored [Nout][Ka] f16)
// grid = (M/16, Nout/16), block = 32 (one wave).
// mode 0: normal store f32 (outf) and/or f16 (outh), ldc row stride, col<Nreal mask
// mode 1: K-head store  f16  [H][Mk][16]   (head = nt)
// mode 2: V-head store  f16  [H][16][Mk]   (head = nt)
// ---------------------------------------------------------------------------
__global__ void gemm16(const _Float16* __restrict__ A, const _Float16* __restrict__ W,
                       const float* __restrict__ bias, float* outf, _Float16* outh,
                       int Ka, int Nreal, int ldc, int mode, int relu, int Mk) {
  const int lane = threadIdx.x;
  const int lo = lane & 15, g = lane >> 4;
  const int mt = blockIdx.x, nt = blockIdx.y;
  const int row = mt * 16 + lo;
  const int col = nt * 16 + lo;

  const _Float16* ap = A + (size_t)row * Ka + g * 8;
  const _Float16* wp = W + (size_t)col * Ka + g * 8;

  v8f acc = zero8f();
  for (int kc = 0; kc < Ka; kc += 32) {
    v16h a = load_a16(ap + kc, ap + kc + 16);        // A: K in 8-blocks per half-wave
    v16h b = load_a16(wp + kc, wp + kc + 16);        // B: K in 8-blocks per half-wave
    acc = __builtin_amdgcn_wmma_f32_16x16x32_f16(false, a, false, b, (short)0, acc, false, false);
  }

  const float bv = (bias && col < Nreal) ? bias[col] : 0.0f;
#pragma unroll
  for (int r = 0; r < 8; ++r) {
    float v = acc[r] + bv;
    if (relu) v = fmaxf(v, 0.0f);
    const int m = mt * 16 + g * 8 + r;               // C layout: M = r + 8*(lane/16)
    if (mode == 0) {
      if (col < Nreal) {
        if (outf) outf[(size_t)m * ldc + col] = v;
        if (outh) outh[(size_t)m * ldc + col] = (_Float16)v;
      }
    } else if (mode == 1) {
      outh[((size_t)nt * Mk + m) * 16 + lo] = (_Float16)v;
    } else {
      outh[((size_t)(nt * 16 + lo)) * Mk + m] = (_Float16)v;
    }
  }
}

// ---------------------------------------------------------------------------
// Flash attention, transposed-logits trick.
// qh: [H][Mq][16] f16, kh: [H][Mk][16] f16, vh: [H][16][Mk] f16
// grid = (ceil(Nq/16), H), block = 32.
// ---------------------------------------------------------------------------
__global__ void attn_kernel(const _Float16* __restrict__ qh, const _Float16* __restrict__ kh,
                            const _Float16* __restrict__ vh, float* outf, _Float16* outh,
                            int Nq, int Mq, int nkeys, int Mk, int cross,
                            const int* __restrict__ viewq, const float* __restrict__ centers,
                            const float* __restrict__ inv2sig2, const float* __restrict__ featpos,
                            int Kpv) {
  const int lane = threadIdx.x;
  const int lo = lane & 15, g = lane >> 4;
  const int qt = blockIdx.x, h = blockIdx.y;
  const int qidx = qt * 16 + lo;                     // query owned by this lane
  const int qv = (qidx < Nq) ? viewq[qidx] : -1;
  float cx = 0.f, cy = 0.f, is2 = 0.f;
  if (cross && qidx < Nq) { cx = centers[2 * qidx]; cy = centers[2 * qidx + 1]; is2 = inv2sig2[qidx]; }

  // q fragment (B operand, hd padded 16->32: elements 0..7 carry K=8g..8g+7, elements 8..15 == zeros)
  v16h bq = zero16();
  {
    v8h t = *(const v8h*)(qh + ((size_t)h * Mq + qt * 16 + lo) * 16 + g * 8);
#pragma unroll
    for (int i = 0; i < 8; ++i) bq[i] = t[i];
  }

  auto biasFor = [&](int key) -> float {
    if (key >= nkeys) return NEG_BIG;
    if (!cross) return (viewq[key] == qv) ? 0.0f : NEG_BIG;
    const int v = key / Kpv;
    const int k = key - v * Kpv;
    const float fx = featpos[2 * k], fy = featpos[2 * k + 1];
    const float dx = cx - fx, dy = cy - fy;
    const float gg = -(dx * dx + dy * dy) * is2;
    float b = (__expf(gg) < EPS32F) ? NEG_BIG : gg;
    if (v != qv) b += NEG_BIG;
    return b;
  };

  float rmax = NEG_BIG, rsum = 0.0f;
  v8f oacc = zero8f();

  for (int kc = 0; kc < nkeys; kc += 32) {
    // key fragments (A operand, hd padded: halves 8..15 zero)
    v16h aL = zero16(), aH = zero16();
    const int key0 = kc + lo, key1 = kc + 16 + lo;
    if (key0 < Mk) {
      v8h t = *(const v8h*)(kh + ((size_t)h * Mk + key0) * 16 + g * 8);
#pragma unroll
      for (int i = 0; i < 8; ++i) aL[i] = t[i];
    }
    if (key1 < Mk) {
      v8h t = *(const v8h*)(kh + ((size_t)h * Mk + key1) * 16 + g * 8);
#pragma unroll
      for (int i = 0; i < 8; ++i) aH[i] = t[i];
    }
    v8f z = zero8f();
    v8f cL = __builtin_amdgcn_wmma_f32_16x16x32_f16(false, aL, false, bq, (short)0, z, false, false);
    v8f cH = __builtin_amdgcn_wmma_f32_16x16x32_f16(false, aH, false, bq, (short)0, z, false, false);

    float l0[8], l1[8];
    float cmax = NEG_BIG;
#pragma unroll
    for (int r = 0; r < 8; ++r) {
      const int kA = kc + g * 8 + r;
      const int kB = kc + 16 + g * 8 + r;
      l0[r] = cL[r] * 0.25f + biasFor(kA);
      l1[r] = cH[r] * 0.25f + biasFor(kB);
      cmax = fmaxf(cmax, fmaxf(l0[r], l1[r]));
    }
    cmax = fmaxf(cmax, __shfl_xor(cmax, 16, 32));
    const float nm = fmaxf(rmax, cmax);
    const float alpha = __expf(rmax - nm);
    float csum = 0.0f;
    v16h pa;                                          // P in exact A-operand layout
#pragma unroll
    for (int r = 0; r < 8; ++r) {
      const float e0 = __expf(l0[r] - nm);
      const float e1 = __expf(l1[r] - nm);
      csum += e0 + e1;
      pa[r] = (_Float16)e0;
      pa[8 + r] = (_Float16)e1;
    }
    csum += __shfl_xor(csum, 16, 32);
    rsum = rsum * alpha + csum;
    rmax = nm;

#pragma unroll
    for (int r = 0; r < 8; ++r) {                     // rescale accumulator per query row
      const float al = __shfl(alpha, r + 8 * g, 32);
      oacc[r] *= al;
    }

    // V fragment (B operand): keys in 8-blocks per half-wave (kc+8g.., kc+16+8g..)
    v16h bv = zero16();
    const int kb = kc + g * 8;
    const _Float16* vp = vh + ((size_t)(h * 16 + lo)) * Mk + kb;
    if (kb + 7 < Mk) {
      v8h t = *(const v8h*)vp;
#pragma unroll
      for (int i = 0; i < 8; ++i) bv[i] = t[i];
    }
    if (kb + 16 + 7 < Mk) {
      v8h t = *(const v8h*)(vp + 16);
#pragma unroll
      for (int i = 0; i < 8; ++i) bv[8 + i] = t[i];
    }

    oacc = __builtin_amdgcn_wmma_f32_16x16x32_f16(false, pa, false, bv, (short)0, oacc, false, false);
  }

#pragma unroll
  for (int r = 0; r < 8; ++r) {
    const float s = __shfl(rsum, r + 8 * g, 32);
    const float val = oacc[r] / fmaxf(s, 1e-30f);
    const int m = qt * 16 + g * 8 + r;
    const int col = h * 16 + lo;
    outf[(size_t)m * C_DIM + col] = val;
    outh[(size_t)m * C_DIM + col] = (_Float16)val;
  }
}

// ---------------------------------------------------------------------------
// LayerNorm(x + dx) * g + b ; writes f32 + f16, zero-pads rows >= Nreal.
// grid = MQ_PAD, block = 128.
// ---------------------------------------------------------------------------
__global__ void ln_res(const float* __restrict__ x, const float* __restrict__ dx,
                       const float* __restrict__ gam, const float* __restrict__ bet,
                       float* outf, _Float16* outh, int Nreal) {
  __shared__ float red[128];
  const int row = blockIdx.x, c = threadIdx.x;
  if (row >= Nreal) { outf[row * 128 + c] = 0.0f; outh[row * 128 + c] = (_Float16)0.0f; return; }
  const float s = x[row * 128 + c] + dx[row * 128 + c];
  red[c] = s; __syncthreads();
  for (int o = 64; o > 0; o >>= 1) { if (c < o) red[c] += red[c + o]; __syncthreads(); }
  const float m = red[0] * (1.0f / 128.0f); __syncthreads();
  const float d = s - m;
  red[c] = d * d; __syncthreads();
  for (int o = 64; o > 0; o >>= 1) { if (c < o) red[c] += red[c + o]; __syncthreads(); }
  const float var = red[0] * (1.0f / 128.0f);
  const float y = d * rsqrtf(var + 1e-5f) * gam[c] + bet[c];
  outf[row * 128 + c] = y;
  outh[row * 128 + c] = (_Float16)y;
}

// weight transpose+convert: dst[n][k] = src[k*stride + coloff + n]  (zero rows >= Nreal)
__global__ void convT(const float* __restrict__ src, _Float16* dst, int Kin, int Nreal,
                      int Nalloc, int stride, int coloff) {
  const int i = blockIdx.x * blockDim.x + threadIdx.x;
  if (i >= Nalloc * Kin) return;
  const int n = i / Kin, k = i - n * Kin;
  dst[i] = (n < Nreal) ? (_Float16)src[(size_t)k * stride + coloff + n] : (_Float16)0.0f;
}

// posembed layer1: relu(pos @ w1 + b1), w1 is [2,128]; zero-pads rows >= Mreal.
__global__ void posembed1(const float* __restrict__ pos, const float* __restrict__ w1,
                          const float* __restrict__ b1, _Float16* out, int Mreal, int Mpad) {
  const int i = blockIdx.x * blockDim.x + threadIdx.x;
  if (i >= Mpad * 128) return;
  const int m = i >> 7, c = i & 127;
  float v = 0.0f;
  if (m < Mreal) v = fmaxf(pos[2 * m] * w1[c] + pos[2 * m + 1] * w1[128 + c] + b1[c], 0.0f);
  out[i] = (_Float16)v;
}

__global__ void add_pe(const float* __restrict__ q, const float* __restrict__ pe,
                       _Float16* out, int Mreal) {
  const int i = blockIdx.x * blockDim.x + threadIdx.x;
  if (i >= MQ_PAD * 128) return;
  const int m = i >> 7;
  out[i] = (m < Mreal) ? (_Float16)(q[i] + pe[i]) : (_Float16)0.0f;
}

// kk[key][c] = feat[v][c][k] + kpe[k][c]   (key = v*K + k)
__global__ void build_kk(const float* __restrict__ feat, const float* __restrict__ kpe,
                         _Float16* kk) {
  const int i = blockIdx.x * blockDim.x + threadIdx.x;
  if (i >= VK_N * C_DIM) return;
  const int key = i >> 7, c = i & 127;
  const int v = key / K_PV, k = key - v * K_PV;
  kk[i] = (_Float16)(feat[((size_t)v * C_DIM + c) * K_PV + k] + kpe[(size_t)k * C_DIM + c]);
}

__global__ void init_q(const float* __restrict__ iqf, float* qf, _Float16* qh16) {
  const int i = blockIdx.x * blockDim.x + threadIdx.x;
  if (i >= MQ_PAD * 128) return;
  const int n = i >> 7, c = i & 127;
  const float v = (n < N_Q) ? iqf[c * N_Q + n] : 0.0f;
  qf[i] = v; qh16[i] = (_Float16)v;
}

__global__ void prep_pos(const float* __restrict__ qpos, const float* __restrict__ kpos,
                         float* cent, float* fpos) {
  const int i = blockIdx.x * blockDim.x + threadIdx.x;
  if (i < 2 * N_Q) {
    cent[i] = sigmf(qpos[i]) * ((i & 1) ? IMG_YC : IMG_XC);
  } else if (i < 2 * N_Q + 2 * K_PV) {
    const int j = i - 2 * N_Q;
    fpos[j] = sigmf(kpos[j]) * ((j & 1) ? IMG_YC : IMG_XC) - 0.5f;
  }
}

__global__ void sigma_k(const float* __restrict__ bbox, float* is2) {
  const int n = blockIdx.x * blockDim.x + threadIdx.x;
  if (n >= N_Q) return;
  const float dx = sigmf(bbox[n * 6 + 2]) * IMG_XC;
  const float dy = sigmf(bbox[n * 6 + 3]) * IMG_YC;
  const float r = ceilf(sqrtf(dx * dx + dy * dy) * 0.5f);
  const float sg = (2.0f * r + 1.0f) / 6.0f;
  is2[n] = 1.0f / (2.0f * sg * sg);
}

__global__ void bbox_init(const float* __restrict__ p6, const float* __restrict__ qpos,
                          float* bbox) {
  const int i = blockIdx.x * blockDim.x + threadIdx.x;
  if (i >= N_Q * 6) return;
  const int n = i / 6, j = i - n * 6;
  bbox[i] = (j < 2) ? qpos[2 * n + j] : p6[i];
}

__global__ void bbox_update(const float* __restrict__ p6, const float* __restrict__ qpos,
                            float* bbox, float* ret) {
  const int i = blockIdx.x * blockDim.x + threadIdx.x;
  if (i >= N_Q * 6) return;
  const int n = i / 6, j = i - n * 6;
  const float t = p6[i] + ((j < 2) ? qpos[2 * n + j] : 0.0f);
  const float s = sigmf(t);
  bbox[i] = s; ret[i] = s;
}

__global__ void inv4_k(const float* __restrict__ l2i, float* invout) {
  const int v = threadIdx.x;
  if (v >= V_N) return;
  const float* m = l2i + v * 16;
  float inv[16];
  inv[0]  =  m[5]*m[10]*m[15] - m[5]*m[11]*m[14] - m[9]*m[6]*m[15] + m[9]*m[7]*m[14] + m[13]*m[6]*m[11] - m[13]*m[7]*m[10];
  inv[4]  = -m[4]*m[10]*m[15] + m[4]*m[11]*m[14] + m[8]*m[6]*m[15] - m[8]*m[7]*m[14] - m[12]*m[6]*m[11] + m[12]*m[7]*m[10];
  inv[8]  =  m[4]*m[9]*m[15]  - m[4]*m[11]*m[13] - m[8]*m[5]*m[15] + m[8]*m[7]*m[13] + m[12]*m[5]*m[11] - m[12]*m[7]*m[9];
  inv[12] = -m[4]*m[9]*m[14]  + m[4]*m[10]*m[13] + m[8]*m[5]*m[14] - m[8]*m[6]*m[13] - m[12]*m[5]*m[10] + m[12]*m[6]*m[9];
  inv[1]  = -m[1]*m[10]*m[15] + m[1]*m[11]*m[14] + m[9]*m[2]*m[15] - m[9]*m[3]*m[14] - m[13]*m[2]*m[11] + m[13]*m[3]*m[10];
  inv[5]  =  m[0]*m[10]*m[15] - m[0]*m[11]*m[14] - m[8]*m[2]*m[15] + m[8]*m[3]*m[14] + m[12]*m[2]*m[11] - m[12]*m[3]*m[10];
  inv[9]  = -m[0]*m[9]*m[15]  + m[0]*m[11]*m[13] + m[8]*m[1]*m[15] - m[8]*m[3]*m[13] - m[12]*m[1]*m[11] + m[12]*m[3]*m[9];
  inv[13] =  m[0]*m[9]*m[14]  - m[0]*m[10]*m[13] - m[8]*m[1]*m[14] + m[8]*m[2]*m[13] + m[12]*m[1]*m[10] - m[12]*m[2]*m[9];
  inv[2]  =  m[1]*m[6]*m[15]  - m[1]*m[7]*m[14]  - m[5]*m[2]*m[15] + m[5]*m[3]*m[14] + m[13]*m[2]*m[7]  - m[13]*m[3]*m[6];
  inv[6]  = -m[0]*m[6]*m[15]  + m[0]*m[7]*m[14]  + m[4]*m[2]*m[15] - m[4]*m[3]*m[14] - m[12]*m[2]*m[7]  + m[12]*m[3]*m[6];
  inv[10] =  m[0]*m[5]*m[15]  - m[0]*m[7]*m[13]  - m[4]*m[1]*m[15] + m[4]*m[3]*m[13] + m[12]*m[1]*m[7]  - m[12]*m[3]*m[5];
  inv[14] = -m[0]*m[5]*m[14]  + m[0]*m[6]*m[13]  + m[4]*m[1]*m[14] - m[4]*m[2]*m[13] - m[12]*m[1]*m[6]  + m[12]*m[2]*m[5];
  inv[3]  = -m[1]*m[6]*m[11]  + m[1]*m[7]*m[10]  + m[5]*m[2]*m[11] - m[5]*m[3]*m[10] - m[9]*m[2]*m[7]   + m[9]*m[3]*m[6];
  inv[7]  =  m[0]*m[6]*m[11]  - m[0]*m[7]*m[10]  - m[4]*m[2]*m[11] + m[4]*m[3]*m[10] + m[8]*m[2]*m[7]   - m[8]*m[3]*m[6];
  inv[11] = -m[0]*m[5]*m[11]  + m[0]*m[7]*m[9]   + m[4]*m[1]*m[11] - m[4]*m[3]*m[9]  - m[8]*m[1]*m[7]   + m[8]*m[3]*m[5];
  inv[15] =  m[0]*m[5]*m[10]  - m[0]*m[6]*m[9]   - m[4]*m[1]*m[10] + m[4]*m[2]*m[9]  + m[8]*m[1]*m[6]   - m[8]*m[2]*m[5];
  float det = m[0]*inv[0] + m[1]*inv[4] + m[2]*inv[8] + m[3]*inv[12];
  det = 1.0f / det;
  float* o = invout + v * 16;
#pragma unroll
  for (int i = 0; i < 16; ++i) o[i] = inv[i] * det;
}

__global__ void pos3d_k(const float* __restrict__ qpos, const int* __restrict__ viewq,
                        const float* __restrict__ invm, float* out2) {
  const int i = blockIdx.x * blockDim.x + threadIdx.x;
  if (i >= N_Q * 32) return;
  const int n = i >> 5, d = i & 31;
  const float px = sigmf(qpos[2 * n]) * IMG_XC * OUT_SFC;
  const float py = sigmf(qpos[2 * n + 1]) * IMG_YC * OUT_SFC;
  const float bin = 53.0f / 1056.0f;                 // (PC3-1)/(D*(1+D))
  const float dep = 1.0f + bin * (float)d * ((float)d + 1.0f);
  const float z = fmaxf(dep, 1e-5f);
  const float c0 = px * z, c1 = py * z, c2 = dep, c3 = 1.0f;
  const float* M = invm + viewq[n] * 16;
#pragma unroll
  for (int r = 0; r < 3; ++r)
    out2[n * 96 + d * 3 + r] = M[r * 4] * c0 + M[r * 4 + 1] * c1 + M[r * 4 + 2] * c2 + M[r * 4 + 3] * c3;
}

__global__ void finalize_k(const float* __restrict__ qf, const float* __restrict__ qpos,
                           float* out) {
  const int i = blockIdx.x * blockDim.x + threadIdx.x;
  if (i < C_DIM * N_Q) {
    const int c = i / N_Q, n = i - c * N_Q;
    out[i] = qf[n * 128 + c];
  } else if (i < C_DIM * N_Q + 2 * N_Q) {
    out[i] = qpos[i - C_DIM * N_Q];
  }
}

// ---------------------------------------------------------------------------
static inline int du(int a, int b) { return (a + b - 1) / b; }

extern "C" void kernel_launch(void* const* d_in, const int* in_sizes, int n_in,
                              void* d_out, int out_size, void* d_ws, size_t ws_size,
                              hipStream_t stream) {
  (void)in_sizes; (void)n_in; (void)out_size; (void)ws_size;
  const float* iqf    = (const float*)d_in[0];
  const float* qpos   = (const float*)d_in[1];
  const float* feat   = (const float*)d_in[2];
  const float* kpos   = (const float*)d_in[3];
  const float* l2i    = (const float*)d_in[4];
  const int*   viewq  = (const int*)d_in[5];
  const float* saqkvw = (const float*)d_in[6];
  const float* saqkvb = (const float*)d_in[7];
  const float* saow   = (const float*)d_in[8];
  const float* saob   = (const float*)d_in[9];
  const float* caqkvw = (const float*)d_in[10];
  const float* caqkvb = (const float*)d_in[11];
  const float* caow   = (const float*)d_in[12];
  const float* caob   = (const float*)d_in[13];
  const float* ln1g = (const float*)d_in[14]; const float* ln1b = (const float*)d_in[15];
  const float* ln2g = (const float*)d_in[16]; const float* ln2b = (const float*)d_in[17];
  const float* ln3g = (const float*)d_in[18]; const float* ln3b = (const float*)d_in[19];
  const float* fw1 = (const float*)d_in[20]; const float* fb1 = (const float*)d_in[21];
  const float* fw2 = (const float*)d_in[22]; const float* fb2 = (const float*)d_in[23];
  const float* qpw1 = (const float*)d_in[24]; const float* qpb1 = (const float*)d_in[25];
  const float* qpw2 = (const float*)d_in[26]; const float* qpb2 = (const float*)d_in[27];
  const float* kpw1 = (const float*)d_in[28]; const float* kpb1 = (const float*)d_in[29];
  const float* kpw2 = (const float*)d_in[30]; const float* kpb2 = (const float*)d_in[31];
  const float* prw1 = (const float*)d_in[32]; const float* prb1 = (const float*)d_in[33];
  const float* prw2 = (const float*)d_in[34]; const float* prb2 = (const float*)d_in[35];
  float* out = (float*)d_out;

  // ---- workspace allocator (deterministic, recomputed per call) ----
  char* base = (char*)d_ws; size_t off = 0;
  auto alloc = [&](size_t bytes) -> void* {
    off = (off + 255) & ~(size_t)255;
    void* p = base + off; off += bytes; return p;
  };
  float*    q_f32 = (float*)alloc(MQ_PAD * 128 * 4);
  _Float16* q_f16 = (_Float16*)alloc(MQ_PAD * 128 * 2);
  _Float16* x16   = (_Float16*)alloc(MQ_PAD * 128 * 2);
  float*    qpe   = (float*)alloc(MQ_PAD * 128 * 4);
  _Float16* pe1   = (_Float16*)alloc((size_t)K_PV * 128 * 2);
  float*    kpe   = (float*)alloc((size_t)K_PV * 128 * 4);
  _Float16* kk16  = (_Float16*)alloc((size_t)VK_N * 128 * 2);
  _Float16* qhB   = (_Float16*)alloc(H_N * MQ_PAD * 16 * 2);
  _Float16* khs   = (_Float16*)alloc(H_N * MQ_PAD * 16 * 2);
  _Float16* vhs   = (_Float16*)alloc(H_N * MQ_PAD * 16 * 2);
  _Float16* khc   = (_Float16*)alloc((size_t)H_N * VK_N * 16 * 2);
  _Float16* vhc   = (_Float16*)alloc((size_t)H_N * VK_N * 16 * 2);
  float*    obuf  = (float*)alloc(MQ_PAD * 128 * 4);
  _Float16* o16   = (_Float16*)alloc(MQ_PAD * 128 * 2);
  float*    aout  = (float*)alloc(MQ_PAD * 128 * 4);
  _Float16* f116  = (_Float16*)alloc(MQ_PAD * 256 * 2);
  _Float16* p116  = (_Float16*)alloc(MQ_PAD * 128 * 2);
  float*    p6    = (float*)alloc(MQ_PAD * 6 * 4);
  float*    bbox  = (float*)alloc(N_Q * 6 * 4);
  float*    is2   = (float*)alloc(N_Q * 4);
  float*    cent  = (float*)alloc(N_Q * 2 * 4);
  float*    fpos  = (float*)alloc(K_PV * 2 * 4);
  _Float16* wA = (_Float16*)alloc(256 * 128 * 2);
  _Float16* wB = (_Float16*)alloc(256 * 128 * 2);
  _Float16* wC = (_Float16*)alloc(256 * 128 * 2);
  _Float16* wD = (_Float16*)alloc(256 * 128 * 2);
  _Float16* pw1T = (_Float16*)alloc(128 * 128 * 2);
  _Float16* pw2T = (_Float16*)alloc(16 * 128 * 2);
  float*    inv6 = (float*)alloc(V_N * 16 * 4);

  const int EB = 256;
  #define G1(n) dim3(du((n), EB)), dim3(EB), 0, stream

  // ---- prologue ----
  init_q<<<G1(MQ_PAD * 128)>>>(iqf, q_f32, q_f16);
  prep_pos<<<G1(2 * N_Q + 2 * K_PV)>>>(qpos, kpos, cent, fpos);
  convT<<<G1(128 * 128)>>>(prw1, pw1T, 128, 128, 128, 128, 0);
  convT<<<G1(16 * 128)>>>(prw2, pw2T, 128, 6, 16, 6, 0);
  // initial pred(q0) -> bbox0
  gemm16<<<dim3(13, 8), dim3(32), 0, stream>>>(q_f16, pw1T, prb1, nullptr, p116, 128, 128, 128, 0, 1, 0);
  gemm16<<<dim3(13, 1), dim3(32), 0, stream>>>(p116, pw2T, prb2, p6, nullptr, 128, 6, 6, 0, 0, 0);
  bbox_init<<<G1(N_Q * 6)>>>(p6, qpos, bbox);

  for (int l = 0; l < 2; ++l) {
    // gaussian sigma from current bbox
    sigma_k<<<G1(N_Q)>>>(bbox, is2);
    // qpe
    posembed1<<<G1(MQ_PAD * 128)>>>(qpos, qpw1 + l * 2 * 128, qpb1 + l * 128, pe1, N_Q, MQ_PAD);
    convT<<<G1(128 * 128)>>>(qpw2 + l * 128 * 128, wA, 128, 128, 128, 128, 0);
    gemm16<<<dim3(13, 8), dim3(32), 0, stream>>>(pe1, wA, qpb2 + l * 128, qpe, nullptr, 128, 128, 128, 0, 0, 0);
    // kpe
    posembed1<<<G1(K_PV * 128)>>>(kpos, kpw1 + l * 2 * 128, kpb1 + l * 128, pe1, K_PV, K_PV);
    convT<<<G1(128 * 128)>>>(kpw2 + l * 128 * 128, wA, 128, 128, 128, 128, 0);
    gemm16<<<dim3(K_PV / 16, 8), dim3(32), 0, stream>>>(pe1, wA, kpb2 + l * 128, kpe, nullptr, 128, 128, 128, 0, 0, 0);

    // ---- self attention ----
    add_pe<<<G1(MQ_PAD * 128)>>>(q_f32, qpe, x16, N_Q);
    convT<<<G1(128 * 128)>>>(saqkvw + l * 128 * 384, wA, 128, 128, 128, 384, 0);
    convT<<<G1(128 * 128)>>>(saqkvw + l * 128 * 384, wB, 128, 128, 128, 384, 128);
    convT<<<G1(128 * 128)>>>(saqkvw + l * 128 * 384, wC, 128, 128, 128, 384, 256);
    convT<<<G1(128 * 128)>>>(saow + l * 128 * 128, wD, 128, 128, 128, 128, 0);
    gemm16<<<dim3(13, 8), dim3(32), 0, stream>>>(x16, wA, saqkvb + l * 384, nullptr, qhB, 128, 128, 0, 1, 0, MQ_PAD);
    gemm16<<<dim3(13, 8), dim3(32), 0, stream>>>(x16, wB, saqkvb + l * 384 + 128, nullptr, khs, 128, 128, 0, 1, 0, MQ_PAD);
    gemm16<<<dim3(13, 8), dim3(32), 0, stream>>>(x16, wC, saqkvb + l * 384 + 256, nullptr, vhs, 128, 128, 0, 2, 0, MQ_PAD);
    attn_kernel<<<dim3(13, 8), dim3(32), 0, stream>>>(qhB, khs, vhs, obuf, o16, N_Q, MQ_PAD,
                                                      N_Q, MQ_PAD, 0, viewq, cent, is2, fpos, K_PV);
    gemm16<<<dim3(13, 8), dim3(32), 0, stream>>>(o16, wD, saob + l * 128, aout, nullptr, 128, 128, 128, 0, 0, 0);
    ln_res<<<dim3(MQ_PAD), dim3(128), 0, stream>>>(q_f32, aout, ln1g + l * 128, ln1b + l * 128, q_f32, q_f16, N_Q);

    // ---- cross attention ----
    add_pe<<<G1(MQ_PAD * 128)>>>(q_f32, qpe, x16, N_Q);
    convT<<<G1(128 * 128)>>>(caqkvw + l * 128 * 384, wA, 128, 128, 128, 384, 0);
    convT<<<G1(128 * 128)>>>(caqkvw + l * 128 * 384, wB, 128, 128, 128, 384, 128);
    convT<<<G1(128 * 128)>>>(caqkvw + l * 128 * 384, wC, 128, 128, 128, 384, 256);
    convT<<<G1(128 * 128)>>>(caow + l * 128 * 128, wD, 128, 128, 128, 128, 0);
    gemm16<<<dim3(13, 8), dim3(32), 0, stream>>>(x16, wA, caqkvb + l * 384, nullptr, qhB, 128, 128, 0, 1, 0, MQ_PAD);
    build_kk<<<G1(VK_N * 128)>>>(feat, kpe, kk16);
    gemm16<<<dim3(VK_N / 16, 8), dim3(32), 0, stream>>>(kk16, wB, caqkvb + l * 384 + 128, nullptr, khc, 128, 128, 0, 1, 0, VK_N);
    gemm16<<<dim3(VK_N / 16, 8), dim3(32), 0, stream>>>(kk16, wC, caqkvb + l * 384 + 256, nullptr, vhc, 128, 128, 0, 2, 0, VK_N);
    attn_kernel<<<dim3(13, 8), dim3(32), 0, stream>>>(qhB, khc, vhc, obuf, o16, N_Q, MQ_PAD,
                                                      VK_N, VK_N, 1, viewq, cent, is2, fpos, K_PV);
    gemm16<<<dim3(13, 8), dim3(32), 0, stream>>>(o16, wD, caob + l * 128, aout, nullptr, 128, 128, 128, 0, 0, 0);
    ln_res<<<dim3(MQ_PAD), dim3(128), 0, stream>>>(q_f32, aout, ln2g + l * 128, ln2b + l * 128, q_f32, q_f16, N_Q);

    // ---- FFN ----
    convT<<<G1(256 * 128)>>>(fw1 + l * 128 * 256, wA, 128, 256, 256, 256, 0);
    gemm16<<<dim3(13, 16), dim3(32), 0, stream>>>(q_f16, wA, fb1 + l * 256, nullptr, f116, 128, 256, 256, 0, 1, 0);
    convT<<<G1(128 * 256)>>>(fw2 + l * 256 * 128, wB, 256, 128, 128, 128, 0);
    gemm16<<<dim3(13, 8), dim3(32), 0, stream>>>(f116, wB, fb2 + l * 128, aout, nullptr, 256, 128, 128, 0, 0, 0);
    ln_res<<<dim3(MQ_PAD), dim3(128), 0, stream>>>(q_f32, aout, ln3g + l * 128, ln3b + l * 128, q_f32, q_f16, N_Q);

    // ---- pred -> bbox, write rets[l] ----
    gemm16<<<dim3(13, 8), dim3(32), 0, stream>>>(q_f16, pw1T, prb1, nullptr, p116, 128, 128, 128, 0, 1, 0);
    gemm16<<<dim3(13, 1), dim3(32), 0, stream>>>(p116, pw2T, prb2, p6, nullptr, 128, 6, 6, 0, 0, 0);
    bbox_update<<<G1(N_Q * 6)>>>(p6, qpos, bbox, out + 45200 + l * 1200);
  }

  // ---- epilogue: q.T, qpos copy, pos3d ----
  finalize_k<<<G1(C_DIM * N_Q + 2 * N_Q)>>>(q_f32, qpos, out);
  inv4_k<<<dim3(1), dim3(32), 0, stream>>>(l2i, inv6);
  pos3d_k<<<G1(N_Q * 32)>>>(qpos, viewq, inv6, out + 26000);
  #undef G1
}
